// MultiHeadAttention_1700807050113
// MI455X (gfx1250) — matrix operand order repaired and checked by reading, not compiled, not run on hardware
//
#include <hip/hip_runtime.h>
#include <hip/hip_bf16.h>

// MI455X / gfx1250 multi-head attention forward.
// B=4, S=2048, E=1024, H=16, Hs=64.
// Matrix math on v_wmma_f32_16x16x32_bf16; global->LDS staging on the CDNA5
// async-to-LDS path (ASYNCcnt) with double-buffered LDS in the GEMMs.
// Fragment loads are batched per k-step so the scheduler can cover DS latency
// with WMMA issue instead of s_wait_dscnt 0 before every matrix op.

#define USE_ASYNC 1

typedef unsigned short u16;
typedef unsigned int   u32;
typedef __attribute__((ext_vector_type(16))) __bf16 v16bf;
typedef __attribute__((ext_vector_type(8)))  float  v8f;

union Frag {            // 16 bf16 operand (A or B matrix half) = 2 x 128-bit loads
  v16bf v;
  uint4 q[2];
};

__device__ __forceinline__ u16 f32_to_bf16(float f) {
  u32 u = __float_as_uint(f);
  u32 r = (u + 0x7FFFu + ((u >> 16) & 1u)) >> 16;   // round-to-nearest-even
  return (u16)r;
}

// 16-byte global -> LDS copy. Async path: GLOBAL_LOAD_ASYNC_TO_LDS_B128
// (ASYNCcnt-tracked, no VGPR round trip). LDS byte offset = low 32 bits of the
// generic pointer (LDS aperture: addr[31:0] is the in-allocation offset).
__device__ __forceinline__ void async_copy_b128(const u16* g, u16* l) {
#if USE_ASYNC
  asm volatile("global_load_async_to_lds_b128 %0, %1, off"
               :: "v"((u32)(size_t)l), "v"(g)
               : "memory");
#else
  *(uint4*)l = *(const uint4*)g;
#endif
}

template <int N>
__device__ __forceinline__ void wait_async() {
#if USE_ASYNC
#if __has_builtin(__builtin_amdgcn_s_wait_asynccnt)
  __builtin_amdgcn_s_wait_asynccnt(N);
#else
  if constexpr (N == 0)      asm volatile("s_wait_asynccnt 0x0" ::: "memory");
  else if constexpr (N == 8) asm volatile("s_wait_asynccnt 0x8" ::: "memory");
  else                       asm volatile("s_wait_asynccnt 0x0" ::: "memory");
#endif
#endif
}

// ---------------------------------------------------------------------------
// fp32 -> bf16 elementwise convert (x)
// ---------------------------------------------------------------------------
__global__ __launch_bounds__(256) void cvt_f32_bf16_kernel(
    const float* __restrict__ in, u16* __restrict__ out, int n4) {
  int i = blockIdx.x * blockDim.x + threadIdx.x;
  if (i >= n4) return;
  float4 f = ((const float4*)in)[i];
  uint2 p;
  p.x = (u32)f32_to_bf16(f.x) | ((u32)f32_to_bf16(f.y) << 16);
  p.y = (u32)f32_to_bf16(f.z) | ((u32)f32_to_bf16(f.w) << 16);
  ((uint2*)out)[i] = p;
}

// ---------------------------------------------------------------------------
// W [K][N] f32  ->  Wt [N][K] bf16  (so GEMM B-fragments read contiguous K)
// ---------------------------------------------------------------------------
__global__ __launch_bounds__(256) void transpose_w_kernel(
    const float* __restrict__ Wq, const float* __restrict__ Wk,
    const float* __restrict__ Wv, const float* __restrict__ Wo,
    u16* __restrict__ Wqt, u16* __restrict__ Wkt,
    u16* __restrict__ Wvt, u16* __restrict__ Wot) {
  const float* W; u16* Wt;
  switch (blockIdx.z) {
    case 0:  W = Wq; Wt = Wqt; break;
    case 1:  W = Wk; Wt = Wkt; break;
    case 2:  W = Wv; Wt = Wvt; break;
    default: W = Wo; Wt = Wot; break;
  }
  __shared__ float tile[32][33];
  int x  = blockIdx.x * 32 + threadIdx.x;   // col n
  int y0 = blockIdx.y * 32;                 // row k
  for (int i = threadIdx.y; i < 32; i += 8)
    tile[i][threadIdx.x] = W[(size_t)(y0 + i) * 1024 + x];
  __syncthreads();
  int x2 = blockIdx.y * 32 + threadIdx.x;   // col k in Wt
  int y2 = blockIdx.x * 32;                 // row n in Wt
  for (int i = threadIdx.y; i < 32; i += 8)
    Wt[(size_t)(y2 + i) * 1024 + x2] = f32_to_bf16(tile[threadIdx.x][i]);
}

// ---------------------------------------------------------------------------
// GEMM: C[M,N] = alpha * A[M,K](bf16) * Bt[N,K](bf16)^T, K=N=1024.
// 256 thr = 8 waves (4 M x 2 N); block tile 128x128, wave tile 32x64.
// k-step 64, double-buffered LDS fed by async-to-LDS loads:
//   issue(buf^1, k+64) -> s_wait_asynccnt(8) -> barrier -> 16 wmma -> barrier
// All 24 fragment ds_loads issued before the 16 wmma so partial dscnt waits
// overlap LDS latency with matrix issue.
// ---------------------------------------------------------------------------
template <bool OUT_BF16>
__global__ __launch_bounds__(256) void gemm_kernel(
    const u16* __restrict__ A, const u16* __restrict__ Bt,
    void* __restrict__ Cout, float alpha) {
  constexpr int K = 1024, N = 1024, KS = 64;
  constexpr int LDSS = 72;                  // 64 + 8 pad (16B align kept, conflicts broken)
  __shared__ u16 As[2][128 * LDSS];
  __shared__ u16 Bs[2][128 * LDSS];

  const int tid  = threadIdx.x;
  const int lane = tid & 31;
  const int w    = tid >> 5;
  const int wm   = w & 3, wn = w >> 2;
  const int m0   = blockIdx.y * 128;
  const int n0   = blockIdx.x * 128;
  const int cl   = lane & 15;
  const int hi   = (lane >> 4) & 1;

  v8f acc[2][4] = {};

  // staging: each thread moves 4 b128 chunks of A and 4 of B per k-step
  const int sr  = tid >> 1;                 // row 0..127
  const int sc0 = (tid & 1) * 32;           // element col base (4 chunks x 8 elems)
  const u16* ga = A  + (size_t)(m0 + sr) * K + sc0;   // hoisted bases
  const u16* gb = Bt + (size_t)(n0 + sr) * K + sc0;
  u16* const la0 = &As[0][sr * LDSS + sc0];
  u16* const lb0 = &Bs[0][sr * LDSS + sc0];
  constexpr int BUFO = 128 * LDSS;          // element offset between buffers

  auto stage = [&](int k0, int buf) {
    const u16* a = ga + k0;
    const u16* b = gb + k0;
    u16* la = la0 + buf * BUFO;
    u16* lb = lb0 + buf * BUFO;
#pragma unroll
    for (int j = 0; j < 4; ++j) {
      async_copy_b128(a + j * 8, la + j * 8);
      async_copy_b128(b + j * 8, lb + j * 8);
    }
  };

  stage(0, 0);
  int buf = 0;
  const int klo = hi ? 16 : 0;              // A-frag: lane<16 K0..15, else K16..31
  const int kb  = hi ? 16 : 0;              // B-frag: lane<16 K0..15, else K16..31
  const u16* const arow = &As[0][(wm * 32 + cl) * LDSS + klo];
  const u16* const brow = &Bs[0][(wn * 64 + cl) * LDSS + kb];

  for (int k0 = 0; k0 < K; k0 += KS) {
    const bool more = (k0 + KS) < K;
    if (more) { stage(k0 + KS, buf ^ 1); wait_async<8>(); }
    else      { wait_async<0>(); }
    __syncthreads();

    // batch-load all fragments for this 64-wide k-step
    Frag af[2][2], bf[2][4];
#pragma unroll
    for (int kc2 = 0; kc2 < 2; ++kc2) {
      const int kc = kc2 * 32;
#pragma unroll
      for (int mt = 0; mt < 2; ++mt) {
        const u16* p = arow + buf * BUFO + mt * 16 * LDSS + kc;
        af[kc2][mt].q[0] = *(const uint4*)p;
        af[kc2][mt].q[1] = *(const uint4*)(p + 8);
      }
#pragma unroll
      for (int nt = 0; nt < 4; ++nt) {
        const u16* p = brow + buf * BUFO + nt * 16 * LDSS + kc;
        bf[kc2][nt].q[0] = *(const uint4*)p;
        bf[kc2][nt].q[1] = *(const uint4*)(p + 8);
      }
    }
#pragma unroll
    for (int kc2 = 0; kc2 < 2; ++kc2)
#pragma unroll
      for (int mt = 0; mt < 2; ++mt)
#pragma unroll
        for (int nt = 0; nt < 4; ++nt)
          acc[mt][nt] = __builtin_amdgcn_wmma_f32_16x16x32_bf16(
              false, af[kc2][mt].v, false, bf[kc2][nt].v, (short)0,
              acc[mt][nt], false, false);
    __syncthreads();
    buf ^= 1;
  }

  // C/D layout: VGPR r, lanes 0-15 -> row r, lanes 16-31 -> row r+8; col = lane%16
  const int roff = hi * 8;
  for (int mt = 0; mt < 2; ++mt)
    for (int nt = 0; nt < 4; ++nt) {
      int row = m0 + wm * 32 + mt * 16 + roff;
      int col = n0 + wn * 64 + nt * 16 + cl;
      for (int r = 0; r < 8; ++r) {
        float v = acc[mt][nt][r] * alpha;
        if constexpr (OUT_BF16)
          ((u16*)Cout)[(size_t)(row + r) * N + col] = f32_to_bf16(v);
        else
          ((float*)Cout)[(size_t)(row + r) * N + col] = v;
      }
    }
}

// ---------------------------------------------------------------------------
// Flash attention: grid (B*H, S/64), block 128 (4 waves, 16 q-rows each).
// Q pre-scaled by (1/sqrt(Hs))*log2(e) so softmax uses exp2.
// ---------------------------------------------------------------------------
__global__ __launch_bounds__(128) void attn_kernel(
    const u16* __restrict__ Qb, const u16* __restrict__ Kb,
    const u16* __restrict__ Vb, u16* __restrict__ Ob) {
  constexpr int LDSS = 72;                  // 64 + 8 pad
  __shared__ u16 Ks[64 * LDSS];             // [key][d]  (row-major == B-layout for QK^T)
  __shared__ u16 Vt[64 * LDSS];             // [d][key]  (transposed == B-layout for PV)
  __shared__ u16 Ps[64 * LDSS];             // per-wave P tiles [4][16][LDSS]

  const int tid  = threadIdx.x;
  const int lane = tid & 31;
  const int w    = tid >> 5;
  const int b    = blockIdx.x >> 4;
  const int h    = blockIdx.x & 15;
  const int q0   = blockIdx.y * 64;         // sequence-local
  const int rb   = b * 2048;                // batch row base
  const int colh = h * 64;
  const int cl   = lane & 15;
  const int hi   = (lane >> 4) & 1;
  const int roff = hi * 8;
  const int klo  = hi ? 16 : 0;             // A-frag K chunk base
  const int kbk  = hi ? 16 : 0;             // B-frag K chunk base

  // Q fragments (16 q-rows x 64 d) -> 2 A-frags, direct from global
  Frag qf[2];
  {
    const u16* qp = Qb + (size_t)(rb + q0 + w * 16 + cl) * 1024 + colh;
#pragma unroll
    for (int j = 0; j < 2; ++j) {
      qf[j].q[0] = *(const uint4*)(qp + j * 32 + klo);
      qf[j].q[1] = *(const uint4*)(qp + j * 32 + klo + 8);
    }
  }

  v8f o[4] = {};
  float m_run[8], l_run[8];
  for (int r = 0; r < 8; ++r) { m_run[r] = -3.0e38f; l_run[r] = 0.f; }
  const v8f zero = {};

  // staging bases
  const int strow = tid >> 1;               // 0..63
  const int stc0  = (tid & 1) * 32;
  const u16* const gk = Kb + (size_t)(rb + strow) * 1024 + colh + stc0;
  u16* const lk = &Ks[strow * LDSS + stc0];

  const int nkb = blockIdx.y + 1;           // causal: only key blocks <= q tile
  for (int kb = 0; kb < nkb; ++kb) {
    const int k0 = kb * 64;
    __syncthreads();                        // previous iteration done reading Ks/Vt

    // stage K tile [key][d] via async-to-LDS: 4 b128 chunks per thread
    {
      const u16* src = gk + (size_t)k0 * 1024;
#pragma unroll
      for (int j = 0; j < 4; ++j) async_copy_b128(src + j * 8, lk + j * 8);
    }
    // stage V tile transposed [d][key]: 4x4 micro-blocks, b64 LDS stores
    for (int bid = tid; bid < 256; bid += 128) {
      const int kk = (bid & 15) * 4;        // key base
      const int dd = (bid >> 4) * 4;        // d base
      uint2 g[4];
#pragma unroll
      for (int i = 0; i < 4; ++i)
        g[i] = *(const uint2*)(Vb + (size_t)(rb + k0 + kk + i) * 1024 + colh + dd);
#pragma unroll
      for (int j = 0; j < 4; ++j) {
        const int sh = (j & 1) * 16;
        ushort4 sv;
        sv.x = (u16)(((j < 2) ? g[0].x : g[0].y) >> sh);
        sv.y = (u16)(((j < 2) ? g[1].x : g[1].y) >> sh);
        sv.z = (u16)(((j < 2) ? g[2].x : g[2].y) >> sh);
        sv.w = (u16)(((j < 2) ? g[3].x : g[3].y) >> sh);
        *(ushort4*)&Vt[(dd + j) * LDSS + kk] = sv;
      }
    }
    wait_async<0>();
    __syncthreads();

    // S = Q * K^T : batch-load all 8 K fragments, then 8 wmma
    Frag kf[4][2];
#pragma unroll
    for (int t = 0; t < 4; ++t) {
      const u16* kp = &Ks[(t * 16 + cl) * LDSS + kbk];
      kf[t][0].q[0] = *(const uint4*)kp;
      kf[t][0].q[1] = *(const uint4*)(kp + 8);
      kf[t][1].q[0] = *(const uint4*)(kp + 32);
      kf[t][1].q[1] = *(const uint4*)(kp + 40);
    }
    v8f s[4];
#pragma unroll
    for (int t = 0; t < 4; ++t) {
      s[t] = zero;
      s[t] = __builtin_amdgcn_wmma_f32_16x16x32_bf16(
          false, qf[0].v, false, kf[t][0].v, (short)0, s[t], false, false);
      s[t] = __builtin_amdgcn_wmma_f32_16x16x32_bf16(
          false, qf[1].v, false, kf[t][1].v, (short)0, s[t], false, false);
    }

    // online softmax (base-2); row stats reduced over 16-lane column groups
    for (int r = 0; r < 8; ++r) {
      int qrow = q0 + w * 16 + r + roff;    // sequence-local row
      float mloc = -3.0e38f;
#pragma unroll
      for (int t = 0; t < 4; ++t) {
        int kcol = k0 + t * 16 + cl;
        if (kcol > qrow) s[t][r] = -3.0e38f;  // causal mask
        mloc = fmaxf(mloc, s[t][r]);
      }
      for (int off = 1; off < 16; off <<= 1)
        mloc = fmaxf(mloc, __shfl_xor(mloc, off, 32));
      float mnew = fmaxf(m_run[r], mloc);
      float corr = exp2f(m_run[r] - mnew);
      m_run[r] = mnew;
      float lsum = 0.f;
#pragma unroll
      for (int t = 0; t < 4; ++t) {
        float p = exp2f(s[t][r] - mnew);
        s[t][r] = p;
        lsum += p;
      }
      for (int off = 1; off < 16; off <<= 1)
        lsum += __shfl_xor(lsum, off, 32);
      l_run[r] = l_run[r] * corr + lsum;
#pragma unroll
      for (int t = 0; t < 4; ++t) o[t][r] *= corr;
      // C-layout -> LDS (row-major P) for A-frag reload
#pragma unroll
      for (int t = 0; t < 4; ++t)
        Ps[(w * 16 + r + roff) * LDSS + t * 16 + cl] = f32_to_bf16(s[t][r]);
    }
    __syncthreads();

    // O += P * V : batch-load P frags + all 8 V frags, then 8 wmma
    Frag pf[2];
    {
      const u16* pp = &Ps[(w * 16 + cl) * LDSS];
#pragma unroll
      for (int j = 0; j < 2; ++j) {
        pf[j].q[0] = *(const uint4*)(pp + j * 32 + klo);
        pf[j].q[1] = *(const uint4*)(pp + j * 32 + klo + 8);
      }
    }
    Frag vf[4][2];
#pragma unroll
    for (int t = 0; t < 4; ++t) {
      const u16* vp = &Vt[(t * 16 + cl) * LDSS + kbk];
      vf[t][0].q[0] = *(const uint4*)vp;
      vf[t][0].q[1] = *(const uint4*)(vp + 8);
      vf[t][1].q[0] = *(const uint4*)(vp + 32);
      vf[t][1].q[1] = *(const uint4*)(vp + 40);
    }
#pragma unroll
    for (int t = 0; t < 4; ++t) {
      o[t] = __builtin_amdgcn_wmma_f32_16x16x32_bf16(
          false, pf[0].v, false, vf[t][0].v, (short)0, o[t], false, false);
      o[t] = __builtin_amdgcn_wmma_f32_16x16x32_bf16(
          false, pf[1].v, false, vf[t][1].v, (short)0, o[t], false, false);
    }
  }

  // finalize: divide by row sum, write bf16 O [M][E]
  for (int t = 0; t < 4; ++t)
    for (int r = 0; r < 8; ++r) {
      int qrow = rb + q0 + w * 16 + r + roff;
      float val = o[t][r] / l_run[r];
      Ob[(size_t)qrow * 1024 + colh + t * 16 + cl] = f32_to_bf16(val);
    }
}

// ---------------------------------------------------------------------------
// Host-side launch
// ---------------------------------------------------------------------------
extern "C" void kernel_launch(void* const* d_in, const int* in_sizes, int n_in,
                              void* d_out, int out_size, void* d_ws, size_t ws_size,
                              hipStream_t stream) {
  (void)in_sizes; (void)n_in; (void)out_size; (void)ws_size;
  const float* x  = (const float*)d_in[0];
  const float* Wq = (const float*)d_in[1];
  const float* Wk = (const float*)d_in[2];
  const float* Wv = (const float*)d_in[3];
  const float* Wo = (const float*)d_in[4];
  float* out = (float*)d_out;

  // workspace layout (bf16 = 2 bytes): x 16MB | 4x Wt 2MB | Q,K,V,O 16MB each
  char* ws = (char*)d_ws;
  const size_t MB = 1024u * 1024u;
  u16* xb  = (u16*)(ws);
  u16* Wqt = (u16*)(ws + 16 * MB);
  u16* Wkt = (u16*)(ws + 18 * MB);
  u16* Wvt = (u16*)(ws + 20 * MB);
  u16* Wot = (u16*)(ws + 22 * MB);
  u16* Qb  = (u16*)(ws + 24 * MB);
  u16* Kb  = (u16*)(ws + 40 * MB);
  u16* Vb  = (u16*)(ws + 56 * MB);
  u16* Ob  = (u16*)(ws + 72 * MB);

  const int M = 4 * 2048;                   // 8192 rows
  const int nx4 = (M * 1024) / 4;           // float4 count for x

  cvt_f32_bf16_kernel<<<nx4 / 256, 256, 0, stream>>>(x, xb, nx4);
  transpose_w_kernel<<<dim3(32, 32, 4), dim3(32, 8), 0, stream>>>(
      Wq, Wk, Wv, Wo, Wqt, Wkt, Wvt, Wot);

  // Q scaled by 1/sqrt(64) * log2(e) so attention softmax runs on exp2
  const float alpha_q = 0.125f * 1.4426950408889634f;
  dim3 ggrid(1024 / 128, M / 128);          // (8, 64)
  gemm_kernel<true><<<ggrid, 256, 0, stream>>>(xb, Wqt, Qb, alpha_q);
  gemm_kernel<true><<<ggrid, 256, 0, stream>>>(xb, Wkt, Kb, 1.0f);
  gemm_kernel<true><<<ggrid, 256, 0, stream>>>(xb, Wvt, Vb, 1.0f);

  attn_kernel<<<dim3(4 * 16, 2048 / 64), 128, 0, stream>>>(Qb, Kb, Vb, Ob);

  gemm_kernel<false><<<ggrid, 256, 0, stream>>>(Ob, Wot, out, 1.0f);
}
